// FeatExtractionEarlyFusion_ball_15341623181557
// MI455X (gfx1250) — matrix operand order repaired and checked by reading, not compiled, not run on hardware
//
#include <hip/hip_runtime.h>
#include <math.h>

#define B_     4
#define N_     2048
#define K_     64
#define FEAT   96
#define GROUPS 8
#define CPG    12           // FEAT / GROUPS
#define EPS_   1e-5f

typedef __attribute__((ext_vector_type(16))) _Float16 v16h;
typedef __attribute__((ext_vector_type(8)))  _Float16 v8h;
typedef __attribute__((ext_vector_type(8)))  float    v8f;
typedef __attribute__((ext_vector_type(4)))  int      v4i;

#ifndef __has_builtin
#define __has_builtin(x) 0
#endif
#if __has_builtin(__builtin_amdgcn_global_load_async_to_lds_b128) && \
    __has_builtin(__builtin_amdgcn_s_wait_asynccnt)
#define HAVE_ASYNC 1
#else
#define HAVE_ASYNC 0
#endif

typedef __attribute__((address_space(1))) v4i* gv4i_p;   // global 128b vector ptr
typedef __attribute__((address_space(3))) v4i* lv4i_p;   // LDS 128b vector ptr

// ---------------------------------------------------------------------------
// WMMA fragment loaders (CDNA5 wave32 layouts, CDNA5 ISA §7.12.2)
// ---------------------------------------------------------------------------
__device__ __forceinline__ v16h load_A_frag(const _Float16* A, int lda,
                                            int m0, int k0, int lane) {
  const int row = m0 + (lane & 15);
  const int kb  = k0 + ((lane >> 4) << 3);
  const v8h lo = *(const v8h*)(A + row * lda + kb);
  const v8h hi = *(const v8h*)(A + row * lda + kb + 16);
  v16h a;
#pragma unroll
  for (int h = 0; h < 8; ++h) { a[h] = lo[h]; a[h + 8] = hi[h]; }
  return a;
}

// B 32x16 f16 from pre-swizzled fragment buffer (lane's 16 halves contiguous)
__device__ __forceinline__ v16h load_B_frag(const _Float16* Wf, int kt, int nt,
                                            int lane) {
  const _Float16* p = Wf + ((size_t)((kt * 6 + nt) * 32 + lane)) * 16;
  const v8h lo = *(const v8h*)(p);
  const v8h hi = *(const v8h*)(p + 8);
  v16h b;
#pragma unroll
  for (int h = 0; h < 8; ++h) { b[h] = lo[h]; b[h + 8] = hi[h]; }
  return b;
}

__device__ __forceinline__ float angle3(float ux, float uy, float uz,
                                        float vx, float vy, float vz) {
  const float cx = uy * vz - uz * vy;
  const float cy = uz * vx - ux * vz;
  const float cz = ux * vy - uy * vx;
  const float cn = sqrtf(cx * cx + cy * cy + cz * cz);
  const float dt = ux * vx + uy * vy + uz * vz;
  return atan2f(cn, dt);
}

// ---------------------------------------------------------------------------
// 0) zero stats accumulators
// ---------------------------------------------------------------------------
__global__ void zero_stats_kernel(float* stats) {
  const int i = threadIdx.x;
  if (i < 3 * B_ * GROUPS * 2) stats[i] = 0.0f;
}

// ---------------------------------------------------------------------------
// 0b) pre-swizzle weights into f16 B-fragment layout (tiny; stays in L2)
// ---------------------------------------------------------------------------
__global__ void prep_weights_kernel(const float* __restrict__ W1,
                                    const float* __restrict__ W2,
                                    const float* __restrict__ W3,
                                    _Float16* wf1, _Float16* wf2, _Float16* wf3) {
  const int w = blockIdx.x;
  const float*  W  = (w == 0) ? W1 : (w == 1) ? W2 : W3;
  _Float16*     Wf = (w == 0) ? wf1 : (w == 1) ? wf2 : wf3;
  const int krows  = (w == 0) ? 32 : FEAT;
  const int kvalid = (w == 0) ? 10 : FEAT;
  for (int i = threadIdx.x; i < krows * FEAT; i += blockDim.x) {
    const int k = i / FEAT, c = i - k * FEAT;
    const int kt = k >> 5, kr = k & 31;
    const int nt = c >> 4, cl = c & 15;
    const int lane = cl + (((kr >> 3) & 1) << 4);
    const int h    = (kr & 7) + ((kr >> 4) << 3);
    const float v  = (k < kvalid) ? W[k * FEAT + c] : 0.0f;
    Wf[((size_t)((kt * 6 + nt) * 32 + lane)) * 16 + h] = (_Float16)v;
  }
}

// ---------------------------------------------------------------------------
// 1) brute-force KNN: workgroup per query, wave32 shuffle argmin selection
// ---------------------------------------------------------------------------
__global__ void __launch_bounds__(256)
knn_kernel(const float* __restrict__ xyz, int* __restrict__ idx_out) {
  __shared__ float dist[N_];
  __shared__ float wv[8];
  __shared__ int   wi[8];

  const int q   = blockIdx.x;
  const int b   = q / N_;
  const int n   = q - b * N_;
  const int tid = threadIdx.x;
  const int lane = tid & 31, wave = tid >> 5;
  const float* bx = xyz + (size_t)b * N_ * 3;

  const float qx = bx[n * 3 + 0], qy = bx[n * 3 + 1], qz = bx[n * 3 + 2];

  for (int p = tid; p < N_; p += 256) {
    const float dx = bx[p * 3 + 0] - qx;
    const float dy = bx[p * 3 + 1] - qy;
    const float dz = bx[p * 3 + 2] - qz;
    dist[p] = dx * dx + dy * dy + dz * dz;
  }
  __syncthreads();

  for (int r = 0; r < K_; ++r) {
    float bv = 3.4e38f; int bi = 0;
#pragma unroll
    for (int j = 0; j < N_ / 256; ++j) {
      const int p = tid + j * 256;
      const float v = dist[p];
      if (v < bv) { bv = v; bi = p; }
    }
#pragma unroll
    for (int s = 16; s > 0; s >>= 1) {       // wave32 min-reduce
      const float ov = __shfl_xor(bv, s, 32);
      const int   oi = __shfl_xor(bi, s, 32);
      if (ov < bv) { bv = ov; bi = oi; }
    }
    if (lane == 0) { wv[wave] = bv; wi[wave] = bi; }
    __syncthreads();
    if (tid == 0) {
      float fv = wv[0]; int fi = wi[0];
#pragma unroll
      for (int w = 1; w < 8; ++w)
        if (wv[w] < fv) { fv = wv[w]; fi = wi[w]; }
      idx_out[(size_t)q * K_ + r] = fi;
      dist[fi] = 3.4e38f;
    }
    __syncthreads();
  }
}

// ---------------------------------------------------------------------------
// 2) fused raw features (xyz,dxyz,ppf) + conv1 (10->96) WMMA + stats
// ---------------------------------------------------------------------------
__global__ void __launch_bounds__(128)
fuse_conv1_kernel(const float* __restrict__ xyz, const float* __restrict__ normals,
                  const int* __restrict__ idx, const _Float16* __restrict__ Wf,
                  const float* __restrict__ b1, _Float16* __restrict__ y_out,
                  float* __restrict__ stats) {
  __shared__ _Float16 Ah[K_][32];        // raw features, K padded 10 -> 32
  __shared__ _Float16 Os[K_][FEAT];      // output staging for wide stores
  __shared__ float    bb[FEAT];
  __shared__ float    lsum[GROUPS], lsq[GROUPS];

  const int q = blockIdx.x, b = q / N_, n = q - b * N_;
  const int tid = threadIdx.x;
  const float* bx = xyz     + (size_t)b * N_ * 3;
  const float* bn = normals + (size_t)b * N_ * 3;

  const float qx = bx[n * 3 + 0], qy = bx[n * 3 + 1], qz = bx[n * 3 + 2];
  const float rx = bn[n * 3 + 0], ry = bn[n * 3 + 1], rz = bn[n * 3 + 2];

  if (tid < GROUPS) { lsum[tid] = 0.0f; lsq[tid] = 0.0f; }
  if (tid < FEAT)   bb[tid] = b1[tid];

  if (tid < K_) {
    const int j  = idx[(size_t)q * K_ + tid];
    const float gx = bx[j * 3 + 0], gy = bx[j * 3 + 1], gz = bx[j * 3 + 2];
    const float ix = bn[j * 3 + 0], iy = bn[j * 3 + 1], iz = bn[j * 3 + 2];
    const float dx = gx - qx, dy = gy - qy, dz = gz - qz;
    const float a0 = angle3(rx, ry, rz, dx, dy, dz);
    const float a1 = angle3(ix, iy, iz, dx, dy, dz);
    const float a2 = angle3(rx, ry, rz, ix, iy, iz);
    const float dn = sqrtf(dx * dx + dy * dy + dz * dz);
    const float f[10] = {qx, qy, qz, dx, dy, dz, a0, a1, a2, dn};
    v8h p0, p1, z;
#pragma unroll
    for (int c = 0; c < 8; ++c) { p0[c] = (_Float16)f[c]; z[c] = (_Float16)0.0f; }
    p1 = z;
    p1[0] = (_Float16)f[8];
    p1[1] = (_Float16)f[9];
    *(v8h*)(&Ah[tid][0])  = p0;
    *(v8h*)(&Ah[tid][8])  = p1;
    *(v8h*)(&Ah[tid][16]) = z;
    *(v8h*)(&Ah[tid][24]) = z;
  }
  __syncthreads();

  const int wave = tid >> 5, lane = tid & 31;
  const int m0 = wave * 16;
  const v16h a = load_A_frag(&Ah[0][0], 32, m0, 0, lane);

  const int cl = lane & 15;
  const int mb = m0 + ((lane >> 4) << 3);

  // issue all 6 WMMAs back-to-back, then post-process (hides D->use hazards)
  v8f acc[6];
#pragma unroll
  for (int nt = 0; nt < 6; ++nt) {
    v8f c0 = {};
    acc[nt] = __builtin_amdgcn_wmma_f32_16x16x32_f16(
        false, a, false, load_B_frag(Wf, 0, nt, lane), (short)0, c0, false, false);
  }
#pragma unroll
  for (int nt = 0; nt < 6; ++nt) {
    const int c    = nt * 16 + cl;
    const float bi = bb[c];
    const int g    = c / CPG;
    float s = 0.0f, s2 = 0.0f;
#pragma unroll
    for (int r = 0; r < 8; ++r) {
      const float y = acc[nt][r] + bi;
      Os[mb + r][c] = (_Float16)y;
      s += y; s2 += y * y;
    }
    atomicAdd(&lsum[g], s);
    atomicAdd(&lsq[g], s2);
  }
  __syncthreads();

  _Float16* yo = y_out + (size_t)q * (K_ * FEAT);
  for (int i = tid; i < (K_ * FEAT) / 8; i += 128)
    *(v8h*)(yo + i * 8) = *(const v8h*)(&Os[0][0] + i * 8);

  if (tid < GROUPS) {
    atomicAdd(&stats[(b * GROUPS + tid) * 2 + 0], lsum[tid]);
    atomicAdd(&stats[(b * GROUPS + tid) * 2 + 1], lsq[tid]);
  }
}

// ---------------------------------------------------------------------------
// 3) fused GroupNorm + ReLU + conv (96->96) WMMA + next-layer stats.
//    Raw y tile is copied global->LDS with async-LDS loads (overlapped with
//    stats/scale computation), reusing the Os staging buffer.
// ---------------------------------------------------------------------------
__global__ void __launch_bounds__(128)
norm_conv_kernel(const _Float16* __restrict__ y_in, const float* __restrict__ stats_in,
                 const float* __restrict__ gamma, const float* __restrict__ beta,
                 const _Float16* __restrict__ Wf, const float* __restrict__ bias,
                 _Float16* __restrict__ y_out, float* __restrict__ stats_out) {
  __shared__ _Float16 Ah[K_][FEAT];      // normalized activations (A matrix)
  __shared__ _Float16 Os[K_][FEAT];      // raw-tile landing zone, then output staging
  __shared__ float    scA[FEAT], shA[FEAT], bb[FEAT];
  __shared__ float    mu[GROUPS], rs[GROUPS];
  __shared__ float    lsum[GROUPS], lsq[GROUPS];

  const int q = blockIdx.x, b = q / N_;
  const int tid = threadIdx.x;
  const _Float16* yb = y_in + (size_t)q * (K_ * FEAT);

#if HAVE_ASYNC
  // async copy raw 12KB tile into LDS while we compute stats/scales
  for (int i = tid; i < (K_ * FEAT) / 8; i += 128)
    __builtin_amdgcn_global_load_async_to_lds_b128(
        (gv4i_p)(v4i*)(yb + i * 8), (lv4i_p)(v4i*)(&Os[0][0] + i * 8), 0, 0);
#else
  __builtin_prefetch(yb + (size_t)tid * 48, 0, 3);
#endif

  if (tid < GROUPS) {
    const float cnt = (float)N_ * (float)K_ * (float)CPG;
    const float s  = stats_in[(b * GROUPS + tid) * 2 + 0];
    const float s2 = stats_in[(b * GROUPS + tid) * 2 + 1];
    const float m  = s / cnt;
    mu[tid] = m;
    rs[tid] = rsqrtf(s2 / cnt - m * m + EPS_);
    lsum[tid] = 0.0f; lsq[tid] = 0.0f;
  }
  __syncthreads();
  if (tid < FEAT) {
    const int g = tid / CPG;
    const float sc = gamma[tid] * rs[g];
    scA[tid] = sc;
    shA[tid] = beta[tid] - mu[g] * sc;
    bb[tid]  = bias[tid];
  }
#if HAVE_ASYNC
  __builtin_amdgcn_s_wait_asynccnt(0);
#endif
  __syncthreads();

  // vectorized normalize + ReLU into A matrix
#if HAVE_ASYNC
  const _Float16* srcp = &Os[0][0];
#else
  const _Float16* srcp = yb;
#endif
  for (int i = tid; i < (K_ * FEAT) / 8; i += 128) {
    const int c0 = (i * 8) % FEAT;
    const v8h v = *(const v8h*)(srcp + i * 8);
    v8h o;
#pragma unroll
    for (int j = 0; j < 8; ++j) {
      const float x = (float)v[j];
      o[j] = (_Float16)fmaxf(x * scA[c0 + j] + shA[c0 + j], 0.0f);
    }
    *(v8h*)(&Ah[0][0] + i * 8) = o;
  }
  __syncthreads();

  const int wave = tid >> 5, lane = tid & 31;
  const int m0 = wave * 16;
  const v16h a0 = load_A_frag(&Ah[0][0], FEAT, m0, 0,  lane);
  const v16h a1 = load_A_frag(&Ah[0][0], FEAT, m0, 32, lane);
  const v16h a2 = load_A_frag(&Ah[0][0], FEAT, m0, 64, lane);

  const int cl = lane & 15;
  const int mb = m0 + ((lane >> 4) << 3);

  // all 18 WMMAs back-to-back, then post-process (hides D->use hazards)
  v8f acc[6];
#pragma unroll
  for (int nt = 0; nt < 6; ++nt) {
    v8f c0 = {};
    c0 = __builtin_amdgcn_wmma_f32_16x16x32_f16(false, a0, false,
            load_B_frag(Wf, 0, nt, lane), (short)0, c0, false, false);
    c0 = __builtin_amdgcn_wmma_f32_16x16x32_f16(false, a1, false,
            load_B_frag(Wf, 1, nt, lane), (short)0, c0, false, false);
    acc[nt] = __builtin_amdgcn_wmma_f32_16x16x32_f16(false, a2, false,
            load_B_frag(Wf, 2, nt, lane), (short)0, c0, false, false);
  }
#pragma unroll
  for (int nt = 0; nt < 6; ++nt) {
    const int c    = nt * 16 + cl;
    const float bi = bb[c];
    const int g    = c / CPG;
    float s = 0.0f, s2 = 0.0f;
#pragma unroll
    for (int r = 0; r < 8; ++r) {
      const float y = acc[nt][r] + bi;
      Os[mb + r][c] = (_Float16)y;
      s += y; s2 += y * y;
    }
    atomicAdd(&lsum[g], s);
    atomicAdd(&lsq[g], s2);
  }
  __syncthreads();

  _Float16* yo = y_out + (size_t)q * (K_ * FEAT);
  for (int i = tid; i < (K_ * FEAT) / 8; i += 128)
    *(v8h*)(yo + i * 8) = *(const v8h*)(&Os[0][0] + i * 8);

  if (tid < GROUPS) {
    atomicAdd(&stats_out[(b * GROUPS + tid) * 2 + 0], lsum[tid]);
    atomicAdd(&stats_out[(b * GROUPS + tid) * 2 + 1], lsq[tid]);
  }
}

// ---------------------------------------------------------------------------
// 4) final GroupNorm + ReLU + max over K neighbors (vectorized)
// ---------------------------------------------------------------------------
__global__ void __launch_bounds__(96)
norm_max_kernel(const _Float16* __restrict__ y_in, const float* __restrict__ stats_in,
                const float* __restrict__ gamma, const float* __restrict__ beta,
                float* __restrict__ out) {
  __shared__ float scA[FEAT], shA[FEAT];
  __shared__ float mu[GROUPS], rs[GROUPS];
  __shared__ float pmax[8][FEAT];

  const int q = blockIdx.x, b = q / N_;
  const int tid = threadIdx.x;               // 0..95
  const _Float16* yb = y_in + (size_t)q * (K_ * FEAT);

  __builtin_prefetch(yb + (size_t)tid * 64, 0, 3);

  if (tid < GROUPS) {
    const float cnt = (float)N_ * (float)K_ * (float)CPG;
    const float s  = stats_in[(b * GROUPS + tid) * 2 + 0];
    const float s2 = stats_in[(b * GROUPS + tid) * 2 + 1];
    const float m  = s / cnt;
    mu[tid] = m;
    rs[tid] = rsqrtf(s2 / cnt - m * m + EPS_);
  }
  __syncthreads();
  {
    const int g = tid / CPG;
    const float sc = gamma[tid] * rs[g];
    scA[tid] = sc;
    shA[tid] = beta[tid] - mu[g] * sc;
  }
  __syncthreads();

  const int j  = tid % 12;                   // which 8-channel vector
  const int kg = tid / 12;                   // row group 0..7
  float m[8];
#pragma unroll
  for (int e = 0; e < 8; ++e) m[e] = 0.0f;
#pragma unroll
  for (int r = 0; r < 8; ++r) {
    const int k = kg * 8 + r;
    const v8h v = *(const v8h*)(yb + k * FEAT + j * 8);
#pragma unroll
    for (int e = 0; e < 8; ++e) {
      const float x = (float)v[e] * scA[j * 8 + e] + shA[j * 8 + e];
      m[e] = fmaxf(m[e], fmaxf(x, 0.0f));
    }
  }
#pragma unroll
  for (int e = 0; e < 8; ++e) pmax[kg][j * 8 + e] = m[e];
  __syncthreads();

  float mx = 0.0f;
#pragma unroll
  for (int g2 = 0; g2 < 8; ++g2) mx = fmaxf(mx, pmax[g2][tid]);
  out[(size_t)q * FEAT + tid] = mx;
}

// ---------------------------------------------------------------------------
extern "C" void kernel_launch(void* const* d_in, const int* in_sizes, int n_in,
                              void* d_out, int out_size, void* d_ws, size_t ws_size,
                              hipStream_t stream) {
  const float* xyz     = (const float*)d_in[0];
  const float* normals = (const float*)d_in[1];
  const float* W1 = (const float*)d_in[2];  const float* b1  = (const float*)d_in[3];
  const float* g1 = (const float*)d_in[4];  const float* be1 = (const float*)d_in[5];
  const float* W2 = (const float*)d_in[6];  const float* b2  = (const float*)d_in[7];
  const float* g2 = (const float*)d_in[8];  const float* be2 = (const float*)d_in[9];
  const float* W3 = (const float*)d_in[10]; const float* b3  = (const float*)d_in[11];
  const float* g3 = (const float*)d_in[12]; const float* be3 = (const float*)d_in[13];

  char* ws = (char*)d_ws;
  size_t off = 0;
  int* idx = (int*)(ws + off);       off += (size_t)B_ * N_ * K_ * sizeof(int);
  off = (off + 255) & ~(size_t)255;
  float* stats = (float*)(ws + off); off += (size_t)3 * B_ * GROUPS * 2 * sizeof(float);
  off = (off + 255) & ~(size_t)255;
  _Float16* wf1 = (_Float16*)(ws + off); off += (size_t)1 * 6 * 32 * 16 * sizeof(_Float16);
  off = (off + 255) & ~(size_t)255;
  _Float16* wf2 = (_Float16*)(ws + off); off += (size_t)3 * 6 * 32 * 16 * sizeof(_Float16);
  off = (off + 255) & ~(size_t)255;
  _Float16* wf3 = (_Float16*)(ws + off); off += (size_t)3 * 6 * 32 * 16 * sizeof(_Float16);
  off = (off + 255) & ~(size_t)255;
  _Float16* y_a = (_Float16*)(ws + off); off += (size_t)B_ * N_ * K_ * FEAT * sizeof(_Float16);
  _Float16* y_b = (_Float16*)(ws + off);

  float* st0 = stats + 0 * B_ * GROUPS * 2;
  float* st1 = stats + 1 * B_ * GROUPS * 2;
  float* st2 = stats + 2 * B_ * GROUPS * 2;

  const int P = B_ * N_;
  zero_stats_kernel<<<1, 256, 0, stream>>>(stats);
  prep_weights_kernel<<<3, 256, 0, stream>>>(W1, W2, W3, wf1, wf2, wf3);
  knn_kernel<<<P, 256, 0, stream>>>(xyz, idx);
  fuse_conv1_kernel<<<P, 128, 0, stream>>>(xyz, normals, idx, wf1, b1, y_a, st0);
  norm_conv_kernel<<<P, 128, 0, stream>>>(y_a, st0, g1, be1, wf2, b2, y_b, st1);
  norm_conv_kernel<<<P, 128, 0, stream>>>(y_b, st1, g2, be2, wf3, b3, y_a, st2);
  norm_max_kernel<<<P, 96, 0, stream>>>(y_a, st2, g3, be3, (float*)d_out);
}
